// GCNHead_1915555414122
// MI455X (gfx1250) — matrix operand order repaired and checked by reading, not compiled, not run on hardware
//
#include <hip/hip_runtime.h>
#include <hip/hip_bf16.h>

typedef __bf16    v16bf __attribute__((ext_vector_type(16)));
typedef float     v8f   __attribute__((ext_vector_type(8)));
typedef unsigned  v8u   __attribute__((ext_vector_type(8)));

// ---------- helpers ----------
__device__ __forceinline__ unsigned fbits(float f) { return __builtin_bit_cast(unsigned, f); }
__device__ __forceinline__ float    bcast(unsigned u) { return __builtin_bit_cast(float, u); }

// truncated bf16 high bits of a float
__device__ __forceinline__ unsigned bf_hi_bits(float x) { return fbits(x) >> 16; }
__device__ __forceinline__ float    bf_hi_f(float x)    { return bcast(fbits(x) & 0xFFFF0000u); }

// pack two floats into one dword of bf16 (truncation); hi part
__device__ __forceinline__ unsigned bfpack_hi(float f0, float f1) {
  return (bf_hi_bits(f1) << 16) | bf_hi_bits(f0);
}
// residual (lo) part
__device__ __forceinline__ unsigned bfpack_lo(float f0, float f1) {
  float r0 = f0 - bf_hi_f(f0), r1 = f1 - bf_hi_f(f1);
  return (bf_hi_bits(r1) << 16) | bf_hi_bits(r0);
}

// monotonic float <-> uint for atomicMax-based segment max
__device__ __forceinline__ unsigned fenc(float f) {
  unsigned u = fbits(f);
  return (u & 0x80000000u) ? ~u : (u | 0x80000000u);
}
__device__ __forceinline__ float fdec(unsigned e) {
  unsigned u = (e & 0x80000000u) ? (e & 0x7FFFFFFFu) : ~e;
  return bcast(u);
}

// split 16 floats into packed bf16 hi/lo pairs (8 dwords each)
__device__ __forceinline__ void pack16(const float* f, unsigned* uh, unsigned* ul) {
#pragma unroll
  for (int v = 0; v < 8; ++v) {
    uh[v] = bfpack_hi(f[2 * v], f[2 * v + 1]);
    ul[v] = bfpack_lo(f[2 * v], f[2 * v + 1]);
  }
}

__device__ __forceinline__ v8f vzero8() {
  v8f z = {0.f, 0.f, 0.f, 0.f, 0.f, 0.f, 0.f, 0.f};
  return z;
}

// c += A*B with bf16 hi/lo split (3 WMMAs ~ fp32 accuracy)
__device__ __forceinline__ v8f wmma3(const unsigned* ah, const unsigned* al,
                                     const unsigned* bh, const unsigned* bl, v8f c) {
  v8u Ah, Al, Bh, Bl;
#pragma unroll
  for (int i = 0; i < 8; ++i) { Ah[i] = ah[i]; Al[i] = al[i]; Bh[i] = bh[i]; Bl[i] = bl[i]; }
  v16bf A  = __builtin_bit_cast(v16bf, Ah);
  v16bf Ao = __builtin_bit_cast(v16bf, Al);
  v16bf B  = __builtin_bit_cast(v16bf, Bh);
  v16bf Bo = __builtin_bit_cast(v16bf, Bl);
  c = __builtin_amdgcn_wmma_f32_16x16x32_bf16(false, A,  false, Bo, (short)0, c, false, false);
  c = __builtin_amdgcn_wmma_f32_16x16x32_bf16(false, Ao, false, B,  (short)0, c, false, false);
  c = __builtin_amdgcn_wmma_f32_16x16x32_bf16(false, A,  false, B,  (short)0, c, false, false);
  return c;
}

// load 16 consecutive-packed floats of an A-run into fa[] via float4
__device__ __forceinline__ void loadA16(const float* base, int kc, int hh, float* fa) {
  const float4 a0 = *(const float4*)(base + kc * 32 + 16 * hh);
  const float4 a1 = *(const float4*)(base + kc * 32 + 16 * hh + 4);
  const float4 a2 = *(const float4*)(base + kc * 32 + 16 * hh + 8);
  const float4 a3 = *(const float4*)(base + kc * 32 + 16 * hh + 12);
  fa[0] = a0.x;  fa[1] = a0.y;  fa[2] = a0.z;  fa[3] = a0.w;
  fa[4] = a1.x;  fa[5] = a1.y;  fa[6] = a1.z;  fa[7] = a1.w;
  fa[8] = a2.x;  fa[9] = a2.y;  fa[10] = a2.z; fa[11] = a2.w;
  fa[12] = a3.x; fa[13] = a3.y; fa[14] = a3.z; fa[15] = a3.w;
}

// ---------- small kernels ----------
__global__ void k_init_deg(float* deg, int n) {
  int i = blockIdx.x * blockDim.x + threadIdx.x;
  if (i < n) deg[i] = 1.0f;  // self-loop contribution
}

__global__ void k_init_pool(unsigned* poolu, int n) {
  int i = blockIdx.x * blockDim.x + threadIdx.x;
  if (i < n) poolu[i] = 0u;  // encodes "below every real value"
}

__global__ void k_deg(const int* __restrict__ ei, float* __restrict__ deg, int E) {
  int i = blockIdx.x * blockDim.x + threadIdx.x;
  if (i < E) atomicAdd(&deg[ei[E + i]], 1.0f);  // dst side
}

__global__ void k_dinv(const float* __restrict__ deg, float* __restrict__ dinv, int n) {
  int i = blockIdx.x * blockDim.x + threadIdx.x;
  if (i < n) {
    float d = deg[i];
    dinv[i] = (d > 0.f) ? rsqrtf(d) : 0.f;
  }
}

// agg[i,:] = h[i,:] * dinv[i]^2   (self-loop term, also serves as zero-init)
__global__ void k_selfinit(const float* __restrict__ h, const float* __restrict__ dinv,
                           float* __restrict__ agg, int n) {
  int i = blockIdx.x * blockDim.x + threadIdx.x;
  if (i < n * 64) {
    int node = i >> 6;
    float di = dinv[node];
    agg[i] = h[i] * di * di;
  }
}

// scatter-add edge messages: one thread = (edge, 4 features)
__global__ void k_edge(const int* __restrict__ ei, const float* __restrict__ h,
                       const float* __restrict__ dinv, float* __restrict__ agg, int E) {
  int g = blockIdx.x * blockDim.x + threadIdx.x;
  if (g >= E * 16) return;
  int e = g >> 4, q = g & 15;
  int s = ei[e], t = ei[E + e];
  float w = dinv[s] * dinv[t];
  const float4 hv = *(const float4*)(h + s * 64 + q * 4);
  float* ap = agg + t * 64 + q * 4;
  atomicAdd(ap + 0, hv.x * w);
  atomicAdd(ap + 1, hv.y * w);
  atomicAdd(ap + 2, hv.z * w);
  atomicAdd(ap + 3, hv.w * w);
}

// bias + leaky_relu(0.2) + segment max into ordered-uint pooled
__global__ void k_pool(const float* __restrict__ agg, const float* __restrict__ bg,
                       const int* __restrict__ batch, unsigned* __restrict__ poolu, int n) {
  int i = blockIdx.x * blockDim.x + threadIdx.x;
  if (i >= n * 64) return;
  int node = i >> 6, d = i & 63;
  float v = agg[i] + bg[d];
  v = (v >= 0.f) ? v : 0.2f * v;
  atomicMax(&poolu[batch[node] * 64 + d], fenc(v));
}

// power iteration for sigma_max(W_fc), 64x64; one block of 64 threads
__global__ void k_sigma(const float* __restrict__ Wfc, float* __restrict__ invsig) {
  __shared__ float v[64], u[64], red[64];
  int t = threadIdx.x;
  v[t] = 1.0f;
  float lam = 1.0f;
  for (int it = 0; it < 100; ++it) {
    __syncthreads();
    float s = 0.f;
    for (int k = 0; k < 64; ++k) s += Wfc[t * 64 + k] * v[k];   // u = W v
    u[t] = s;
    __syncthreads();
    float w = 0.f;
    for (int o = 0; o < 64; ++o) w += Wfc[o * 64 + t] * u[o];   // w = W^T u
    red[t] = v[t] * w;                                          // lambda = v . w (v unit)
    __syncthreads();
    for (int st = 32; st > 0; st >>= 1) { if (t < st) red[t] += red[t + st]; __syncthreads(); }
    lam = red[0];
    __syncthreads();
    red[t] = w * w;
    __syncthreads();
    for (int st = 32; st > 0; st >>= 1) { if (t < st) red[t] += red[t + st]; __syncthreads(); }
    float nrm2 = red[0];
    __syncthreads();
    v[t] = w * rsqrtf(fmaxf(nrm2, 1e-30f));
  }
  if (t == 0) invsig[0] = rsqrtf(fmaxf(lam, 1e-30f));  // 1/sigma = 1/sqrt(lambda_max)
}

// ---------- WMMA GEMM: h = x @ W_gcn  (N x 64 @ 64 x 64) ----------
// LDS holds W pre-packed in the exact B-fragment register layout:
//   sW[n*32 + p] = pack_bf16(W[2p][n], W[2p+1][n]),  n = column, p = K-pair
// so a lane's B fragment (chunk kc, half hh) is 8 consecutive dwords.
__global__ void k_gemm_xw(const float* __restrict__ x, const float* __restrict__ Wg,
                          float* __restrict__ h, int ntiles) {
  __shared__ unsigned sWh[64 * 32];
  __shared__ unsigned sWl[64 * 32];
  int t = threadIdx.x;  // 256 threads, 8 waves
#pragma unroll
  for (int i = 0; i < 8; ++i) {
    int s = i * 256 + t;        // 2048 pack slots
    int n = s & 63, p = s >> 6; // column, K-pair
    float w0 = Wg[(2 * p) * 64 + n];
    float w1 = Wg[(2 * p + 1) * 64 + n];
    sWh[n * 32 + p] = bfpack_hi(w0, w1);
    sWl[n * 32 + p] = bfpack_lo(w0, w1);
  }
  __syncthreads();

  int wave = t >> 5, lane = t & 31;
  int tile = blockIdx.x * 8 + wave;
  if (tile >= ntiles) return;  // wave-uniform; EXEC stays all-ones for WMMA

  int m = lane & 15, hh = lane >> 4;  // A row / B column = m; K half = hh
  const float* xr = x + (size_t)(tile * 16 + m) * 64;

  v8f cacc[4];
#pragma unroll
  for (int i = 0; i < 4; ++i) cacc[i] = vzero8();

#pragma unroll
  for (int kc = 0; kc < 2; ++kc) {
    float fa[16];
    loadA16(xr, kc, hh, fa);
    unsigned ah[8], al[8];
    pack16(fa, ah, al);

#pragma unroll
    for (int nt = 0; nt < 4; ++nt) {
      int n = nt * 16 + m;
      const unsigned* bph = &sWh[n * 32 + kc * 16 + 8 * hh];
      const unsigned* bpl = &sWl[n * 32 + kc * 16 + 8 * hh];
      uint4 h0 = *(const uint4*)(bph);
      uint4 h1 = *(const uint4*)(bph + 4);
      uint4 l0 = *(const uint4*)(bpl);
      uint4 l1 = *(const uint4*)(bpl + 4);
      unsigned bh[8] = {h0.x, h0.y, h0.z, h0.w, h1.x, h1.y, h1.z, h1.w};
      unsigned bl[8] = {l0.x, l0.y, l0.z, l0.w, l1.x, l1.y, l1.z, l1.w};
      cacc[nt] = wmma3(ah, al, bh, bl, cacc[nt]);
    }
  }

  int row0 = tile * 16;
#pragma unroll
  for (int nt = 0; nt < 4; ++nt) {
#pragma unroll
    for (int r = 0; r < 8; ++r) {
      int M = r + 8 * hh;
      h[(size_t)(row0 + M) * 64 + nt * 16 + m] = cacc[nt][r];
    }
  }
}

// ---------- WMMA GEMM: out = pooled @ (W_fc/sigma)^T + b_fc  (64x64) ----------
__global__ void k_final(const unsigned* __restrict__ poolu, const float* __restrict__ Wfc,
                        const float* __restrict__ bfc, const float* __restrict__ invsig,
                        float* __restrict__ out) {
  __shared__ float sP[64 * 64];
  int t = threadIdx.x;  // 128 threads, 4 waves
  float is = invsig[0];
#pragma unroll
  for (int i = 0; i < 32; ++i) {
    int idx = i * 128 + t;
    sP[idx] = fdec(poolu[idx]);
  }
  __syncthreads();

  int wave = t >> 5, lane = t & 31;
  int m = lane & 15, hh = lane >> 4;
  int mt = wave;  // row tile of pooled
  const float* pr = &sP[(mt * 16 + m) * 64];

  v8f cacc[4];
#pragma unroll
  for (int i = 0; i < 4; ++i) cacc[i] = vzero8();

#pragma unroll
  for (int kc = 0; kc < 2; ++kc) {
    float fa[16];
    loadA16(pr, kc, hh, fa);
    unsigned ah[8], al[8];
    pack16(fa, ah, al);

#pragma unroll
    for (int nt = 0; nt < 4; ++nt) {
      // B[k][n] = W_fc[n][k] / sigma ; 16 consecutive k -> contiguous row segment
      const float* wr = Wfc + (nt * 16 + m) * 64 + kc * 32 + 16 * hh;
      const float4 w0 = *(const float4*)(wr);
      const float4 w1 = *(const float4*)(wr + 4);
      const float4 w2 = *(const float4*)(wr + 8);
      const float4 w3 = *(const float4*)(wr + 12);
      float fb[16] = {w0.x * is, w0.y * is, w0.z * is, w0.w * is,
                      w1.x * is, w1.y * is, w1.z * is, w1.w * is,
                      w2.x * is, w2.y * is, w2.z * is, w2.w * is,
                      w3.x * is, w3.y * is, w3.z * is, w3.w * is};
      unsigned bh[8], bl[8];
      pack16(fb, bh, bl);
      cacc[nt] = wmma3(ah, al, bh, bl, cacc[nt]);
    }
  }

#pragma unroll
  for (int nt = 0; nt < 4; ++nt) {
    float bias = bfc[nt * 16 + m];
#pragma unroll
    for (int r = 0; r < 8; ++r) {
      int M = r + 8 * hh;
      out[(mt * 16 + M) * 64 + nt * 16 + m] = cacc[nt][r] + bias;
    }
  }
}

// ---------- launch ----------
extern "C" void kernel_launch(void* const* d_in, const int* in_sizes, int n_in,
                              void* d_out, int out_size, void* d_ws, size_t ws_size,
                              hipStream_t stream) {
  (void)n_in; (void)out_size; (void)ws_size;
  const float* x    = (const float*)d_in[0];
  const float* Wg   = (const float*)d_in[1];
  const float* bg   = (const float*)d_in[2];
  const float* Wfc  = (const float*)d_in[3];
  const float* bfc  = (const float*)d_in[4];
  const int*   ei   = (const int*)d_in[5];
  const int*   bat  = (const int*)d_in[6];

  const int N = in_sizes[0] / 64;
  const int E = in_sizes[5] / 2;

  char* ws = (char*)d_ws;
  float*    deg    = (float*)ws;
  float*    dinv   = (float*)(ws + (size_t)N * 4);
  float*    h      = (float*)(ws + (size_t)N * 8);
  float*    agg    = (float*)(ws + (size_t)N * 8 + (size_t)N * 256);
  unsigned* poolu  = (unsigned*)(ws + (size_t)N * 8 + (size_t)N * 512);
  float*    invsig = (float*)(ws + (size_t)N * 8 + (size_t)N * 512 + 64 * 64 * 4);
  float*    out    = (float*)d_out;

  const int ntiles = N / 16;

  k_init_deg <<<(N + 255) / 256, 256, 0, stream>>>(deg, N);
  k_init_pool<<<(64 * 64 + 255) / 256, 256, 0, stream>>>(poolu, 64 * 64);
  k_deg      <<<(E + 255) / 256, 256, 0, stream>>>(ei, deg, E);
  k_dinv     <<<(N + 255) / 256, 256, 0, stream>>>(deg, dinv, N);
  k_gemm_xw  <<<(ntiles + 7) / 8, 256, 0, stream>>>(x, Wg, h, ntiles);
  k_selfinit <<<(N * 64 + 255) / 256, 256, 0, stream>>>(h, dinv, agg, N);
  k_edge     <<<(E * 16 + 255) / 256, 256, 0, stream>>>(ei, h, dinv, agg, E);
  k_pool     <<<(N * 64 + 255) / 256, 256, 0, stream>>>(agg, bg, bat, poolu, N);
  k_sigma    <<<1, 64, 0, stream>>>(Wfc, invsig);
  k_final    <<<1, 128, 0, stream>>>(poolu, Wfc, bfc, invsig, out);
}
